// WaveNetNvidia_16372415332781
// MI455X (gfx1250) — matrix operand order repaired and checked by reading, not compiled, not run on hardware
//
#include <hip/hip_runtime.h>

// ---------------------------------------------------------------------------
// Types for CDNA5 WMMA (wave32): v_wmma_f32_16x16x32_bf16
// ---------------------------------------------------------------------------
typedef __attribute__((ext_vector_type(16))) __bf16 v16bf;
typedef __attribute__((ext_vector_type(8)))  __bf16 v8bf;
typedef __attribute__((ext_vector_type(8)))  float  v8f;

#define M_TOT 32768   // 8 * 4096 tokens
#define T_LEN 4096
#define NCH   256
#define SPC   640
#define NL    12
#define MSUB  4       // 16-row subtiles per wave (wave owns 64 rows)

union AFrag { v16bf v; v8bf h[2]; };

__device__ __forceinline__ v8f wmma_bf16(v16bf a, v16bf b, v8f c) {
  // (neg_a, A, neg_b, B, c_mod, C, reuse_a, reuse_b)
  return __builtin_amdgcn_wmma_f32_16x16x32_bf16(false, a, false, b, (short)0, c,
                                                 false, false);
}

__device__ __forceinline__ v16bf load_afrag(const __bf16* row, int k0, int half) {
  AFrag u;
  u.h[0] = *(const v8bf*)(row + k0 + half * 16);
  u.h[1] = *(const v8bf*)(row + k0 + half * 16 + 8);
  return u.v;
}

// ---------------------------------------------------------------------------
// Prep kernels
// ---------------------------------------------------------------------------
__global__ void k_cvt_bf16(const float* __restrict__ in, __bf16* __restrict__ out,
                           long total) {
  long i = (long)blockIdx.x * blockDim.x + threadIdx.x;
  if (i < total) out[i] = (__bf16)in[i];
}

// [L][C][D] fp32  ->  [L][D][C] bf16   (weights to N-major / K-contiguous)
__global__ void k_transpose_cvt(const float* __restrict__ in, __bf16* __restrict__ out,
                                int L, int C, int D) {
  long i = (long)blockIdx.x * blockDim.x + threadIdx.x;
  long total = (long)L * C * D;
  if (i >= total) return;
  int l = (int)(i / ((long)C * D));
  int rem = (int)(i - (long)l * C * D);
  int c = rem / D;
  int d = rem - c * D;
  out[((size_t)l * D + d) * C + c] = (__bf16)in[i];
}

__global__ void k_bias_add(const float* __restrict__ a, const float* __restrict__ b,
                           float* __restrict__ out, int total) {
  int i = blockIdx.x * blockDim.x + threadIdx.x;
  if (i < total) out[i] = a[i] + b[i];
}

// started = audio @ w_start + b_start   (K = 4)
__global__ void k_start(const float* __restrict__ audio,    // [M_TOT][4]
                        const float* __restrict__ w_start,  // [4][256]
                        const float* __restrict__ b_start,  // [256]
                        float* __restrict__ started_f,
                        __bf16* __restrict__ started_bf) {
  int i = blockIdx.x * blockDim.x + threadIdx.x;
  if (i >= M_TOT * NCH) return;
  int row = i >> 8;
  int c = i & 255;
  float s = b_start[c];
#pragma unroll
  for (int k = 0; k < 4; ++k) s += audio[row * 4 + k] * w_start[k * NCH + c];
  started_f[i] = s;
  started_bf[i] = (__bf16)s;
}

// out = skip @ w_end + b_end   (N = 8)
__global__ void k_end(const float* __restrict__ skip_f,  // [M_TOT][256]
                      const float* __restrict__ w_end,   // [256][8]
                      const float* __restrict__ b_end,   // [8]
                      float* __restrict__ out) {
  int i = blockIdx.x * blockDim.x + threadIdx.x;
  if (i >= M_TOT * 8) return;
  int row = i >> 3;
  int j = i & 7;
  float s = b_end[j];
  const float* r = skip_f + (size_t)row * NCH;
  for (int k = 0; k < NCH; ++k) s += r[k] * w_end[k * 8 + j];
  out[i] = s;
}

// ---------------------------------------------------------------------------
// Gate kernel: acts = dilated_conv(started, w_in) + spect @ w_cond + bias
//              activated = tanh(acts[:, :256]) * sigmoid(acts[:, 256:])
// Block = 8 waves, block owns a 64-row tile. Wave w owns gate cols
// [32w, 32w+32) paired with sigmoid cols [+256), for 4 M-subtiles of 16 rows
// => 16 accumulators, per K-step: 4 A-loads + 4 B-loads feed 16 WMMAs.
// SAME-padding handled branch-free via a zeroed guard row.
// ---------------------------------------------------------------------------
__global__ __launch_bounds__(256) void k_gate(
    const __bf16* __restrict__ started_bf,  // [M_TOT][256]
    const __bf16* __restrict__ spect_bf,    // [M_TOT][640]
    const __bf16* __restrict__ w_in_T,      // [3][512][256]  (this layer)
    const __bf16* __restrict__ w_cond_T,    // [512][640]     (this layer)
    const float*  __restrict__ bias_g,      // [512]  b_in + b_cond
    const __bf16* __restrict__ zguard,      // [256] zeros
    __bf16* __restrict__ act_bf,            // [M_TOT][256]
    int dilation) {
  const int lane = threadIdx.x & 31;
  const int wave = threadIdx.x >> 5;
  const int half = lane >> 4;   // 0: lanes 0-15, 1: lanes 16-31
  const int l15  = lane & 15;
  const int mbase = blockIdx.x * (16 * MSUB);
  const int t0 = mbase & (T_LEN - 1);   // position within sample (64-aligned)
  const int bbase = mbase - t0;         // sample base row
  const int cb = wave * 32;             // gate column base [0,256)
  const int bcol = half * 16;           // B fragment K sub-offset (halves)

  v8f acc[MSUB][4];   // [subtile][t0,t1,s0,s1]
  {
    float bt0 = bias_g[cb + l15];
    float bt1 = bias_g[cb + 16 + l15];
    float bs0 = bias_g[cb + 256 + l15];
    float bs1 = bias_g[cb + 272 + l15];
#pragma unroll
    for (int m = 0; m < MSUB; ++m)
#pragma unroll
      for (int j = 0; j < 8; ++j) {
        acc[m][0][j] = bt0; acc[m][1][j] = bt1;
        acc[m][2][j] = bs0; acc[m][3][j] = bs1;
      }
  }

  // --- 3 conv taps over started (K = 256 each, shifted rows, SAME pad) ---
#pragma unroll
  for (int tap = 0; tap < 3; ++tap) {
    const __bf16* arow[MSUB];
#pragma unroll
    for (int m = 0; m < MSUB; ++m) {
      int tr = t0 + 16 * m + l15 + (tap - 1) * dilation;
      bool valid = (tr >= 0) && (tr < T_LEN);
      arow[m] = valid ? (started_bf + (size_t)(bbase + tr) * NCH) : zguard;
    }
    const __bf16* wb = w_in_T + (size_t)tap * 512 * NCH;
    const __bf16* b0 = wb + (size_t)(cb       + l15) * NCH + bcol;
    const __bf16* b1 = wb + (size_t)(cb + 16  + l15) * NCH + bcol;
    const __bf16* b2 = wb + (size_t)(cb + 256 + l15) * NCH + bcol;
    const __bf16* b3 = wb + (size_t)(cb + 272 + l15) * NCH + bcol;
    for (int k0 = 0; k0 < NCH; k0 += 32) {
      v16bf B0 = *(const v16bf*)(b0 + k0);
      v16bf B1 = *(const v16bf*)(b1 + k0);
      v16bf B2 = *(const v16bf*)(b2 + k0);
      v16bf B3 = *(const v16bf*)(b3 + k0);
#pragma unroll
      for (int m = 0; m < MSUB; ++m) {
        v16bf A = load_afrag(arow[m], k0, half);
        acc[m][0] = wmma_bf16(A, B0, acc[m][0]);
        acc[m][1] = wmma_bf16(A, B1, acc[m][1]);
        acc[m][2] = wmma_bf16(A, B2, acc[m][2]);
        acc[m][3] = wmma_bf16(A, B3, acc[m][3]);
      }
    }
  }

  // --- conditioning: spect @ w_cond  (K = 640, L2-resident spect) ---
  {
    const __bf16* srow[MSUB];
#pragma unroll
    for (int m = 0; m < MSUB; ++m)
      srow[m] = spect_bf + (size_t)(mbase + 16 * m + l15) * SPC;
    const __bf16* b0 = w_cond_T + (size_t)(cb       + l15) * SPC + bcol;
    const __bf16* b1 = w_cond_T + (size_t)(cb + 16  + l15) * SPC + bcol;
    const __bf16* b2 = w_cond_T + (size_t)(cb + 256 + l15) * SPC + bcol;
    const __bf16* b3 = w_cond_T + (size_t)(cb + 272 + l15) * SPC + bcol;
    for (int k0 = 0; k0 < SPC; k0 += 32) {
      v16bf B0 = *(const v16bf*)(b0 + k0);
      v16bf B1 = *(const v16bf*)(b1 + k0);
      v16bf B2 = *(const v16bf*)(b2 + k0);
      v16bf B3 = *(const v16bf*)(b3 + k0);
#pragma unroll
      for (int m = 0; m < MSUB; ++m) {
        v16bf A = load_afrag(srow[m], k0, half);
        acc[m][0] = wmma_bf16(A, B0, acc[m][0]);
        acc[m][1] = wmma_bf16(A, B1, acc[m][1]);
        acc[m][2] = wmma_bf16(A, B2, acc[m][2]);
        acc[m][3] = wmma_bf16(A, B3, acc[m][3]);
      }
    }
  }

  // --- gated activation epilogue, store bf16 ---
#pragma unroll
  for (int m = 0; m < MSUB; ++m)
#pragma unroll
    for (int j = 0; j < 8; ++j) {
      int row = mbase + 16 * m + j + half * 8;
      float g0 = tanhf(acc[m][0][j]) * (1.0f / (1.0f + expf(-acc[m][2][j])));
      float g1 = tanhf(acc[m][1][j]) * (1.0f / (1.0f + expf(-acc[m][3][j])));
      act_bf[(size_t)row * NCH + cb + l15]      = (__bf16)g0;
      act_bf[(size_t)row * NCH + cb + 16 + l15] = (__bf16)g1;
    }
}

// ---------------------------------------------------------------------------
// Res kernel (layers 0..10): rs = activated @ w_res + b_res;
// started += rs[:, :256] (fp32 master + bf16 copy); skip (+)= rs[:, 256:]
// ---------------------------------------------------------------------------
__global__ __launch_bounds__(256) void k_res(
    const __bf16* __restrict__ act_bf,   // [M_TOT][256]
    const __bf16* __restrict__ w_res_T,  // [512][256]
    const float*  __restrict__ bres,     // [512]
    float* __restrict__ started_f,
    __bf16* __restrict__ started_bf,
    float* __restrict__ skip_f,
    int first) {
  const int lane = threadIdx.x & 31;
  const int wave = threadIdx.x >> 5;
  const int half = lane >> 4;
  const int l15  = lane & 15;
  const int mbase = blockIdx.x * (16 * MSUB);
  const int cb = wave * 32;
  const int bcol = half * 16;

  v8f acc[MSUB][4];   // [subtile][r0,r1,s0,s1]
  {
    float br0 = bres[cb + l15];
    float br1 = bres[cb + 16 + l15];
    float bs0 = bres[cb + 256 + l15];
    float bs1 = bres[cb + 272 + l15];
#pragma unroll
    for (int m = 0; m < MSUB; ++m)
#pragma unroll
      for (int j = 0; j < 8; ++j) {
        acc[m][0][j] = br0; acc[m][1][j] = br1;
        acc[m][2][j] = bs0; acc[m][3][j] = bs1;
      }
  }

  const __bf16* arow[MSUB];
#pragma unroll
  for (int m = 0; m < MSUB; ++m)
    arow[m] = act_bf + (size_t)(mbase + 16 * m + l15) * NCH;
  const __bf16* b0 = w_res_T + (size_t)(cb       + l15) * NCH + bcol;
  const __bf16* b1 = w_res_T + (size_t)(cb + 16  + l15) * NCH + bcol;
  const __bf16* b2 = w_res_T + (size_t)(cb + 256 + l15) * NCH + bcol;
  const __bf16* b3 = w_res_T + (size_t)(cb + 272 + l15) * NCH + bcol;
  for (int k0 = 0; k0 < NCH; k0 += 32) {
    v16bf B0 = *(const v16bf*)(b0 + k0);
    v16bf B1 = *(const v16bf*)(b1 + k0);
    v16bf B2 = *(const v16bf*)(b2 + k0);
    v16bf B3 = *(const v16bf*)(b3 + k0);
#pragma unroll
    for (int m = 0; m < MSUB; ++m) {
      v16bf A = load_afrag(arow[m], k0, half);
      acc[m][0] = wmma_bf16(A, B0, acc[m][0]);
      acc[m][1] = wmma_bf16(A, B1, acc[m][1]);
      acc[m][2] = wmma_bf16(A, B2, acc[m][2]);
      acc[m][3] = wmma_bf16(A, B3, acc[m][3]);
    }
  }

#pragma unroll
  for (int m = 0; m < MSUB; ++m)
#pragma unroll
    for (int j = 0; j < 8; ++j) {
      int row = mbase + 16 * m + j + half * 8;
      size_t i0 = (size_t)row * NCH + cb + l15;
      size_t i1 = i0 + 16;
      float ns0 = started_f[i0] + acc[m][0][j];
      float ns1 = started_f[i1] + acc[m][1][j];
      started_f[i0] = ns0; started_f[i1] = ns1;
      started_bf[i0] = (__bf16)ns0; started_bf[i1] = (__bf16)ns1;
      if (first) { skip_f[i0] = acc[m][2][j];  skip_f[i1] = acc[m][3][j]; }
      else       { skip_f[i0] += acc[m][2][j]; skip_f[i1] += acc[m][3][j]; }
    }
}

// ---------------------------------------------------------------------------
// Last layer: skip += activated @ w_res_last + b_res_last  (N = 256)
// ---------------------------------------------------------------------------
__global__ __launch_bounds__(256) void k_last(
    const __bf16* __restrict__ act_bf,  // [M_TOT][256]
    const __bf16* __restrict__ w_rl_T,  // [256][256]
    const float*  __restrict__ brl,     // [256]
    float* __restrict__ skip_f) {
  const int lane = threadIdx.x & 31;
  const int wave = threadIdx.x >> 5;
  const int half = lane >> 4;
  const int l15  = lane & 15;
  const int mbase = blockIdx.x * (16 * MSUB);
  const int cb = wave * 32;
  const int bcol = half * 16;

  v8f acc[MSUB][2];
  {
    float c0 = brl[cb + l15];
    float c1 = brl[cb + 16 + l15];
#pragma unroll
    for (int m = 0; m < MSUB; ++m)
#pragma unroll
      for (int j = 0; j < 8; ++j) { acc[m][0][j] = c0; acc[m][1][j] = c1; }
  }

  const __bf16* arow[MSUB];
#pragma unroll
  for (int m = 0; m < MSUB; ++m)
    arow[m] = act_bf + (size_t)(mbase + 16 * m + l15) * NCH;
  const __bf16* b0 = w_rl_T + (size_t)(cb      + l15) * NCH + bcol;
  const __bf16* b1 = w_rl_T + (size_t)(cb + 16 + l15) * NCH + bcol;
  for (int k0 = 0; k0 < NCH; k0 += 32) {
    v16bf B0 = *(const v16bf*)(b0 + k0);
    v16bf B1 = *(const v16bf*)(b1 + k0);
#pragma unroll
    for (int m = 0; m < MSUB; ++m) {
      v16bf A = load_afrag(arow[m], k0, half);
      acc[m][0] = wmma_bf16(A, B0, acc[m][0]);
      acc[m][1] = wmma_bf16(A, B1, acc[m][1]);
    }
  }

#pragma unroll
  for (int m = 0; m < MSUB; ++m)
#pragma unroll
    for (int j = 0; j < 8; ++j) {
      int row = mbase + 16 * m + j + half * 8;
      size_t i0 = (size_t)row * NCH + cb + l15;
      skip_f[i0]      += acc[m][0][j];
      skip_f[i0 + 16] += acc[m][1][j];
    }
}

// ---------------------------------------------------------------------------
// Host launcher
// ---------------------------------------------------------------------------
extern "C" void kernel_launch(void* const* d_in, const int* in_sizes, int n_in,
                              void* d_out, int out_size, void* d_ws, size_t ws_size,
                              hipStream_t stream) {
  const float* audio      = (const float*)d_in[0];
  const float* spect      = (const float*)d_in[1];
  const float* w_start    = (const float*)d_in[2];
  const float* b_start    = (const float*)d_in[3];
  const float* w_in       = (const float*)d_in[4];
  const float* b_in       = (const float*)d_in[5];
  const float* w_cond     = (const float*)d_in[6];
  const float* b_cond     = (const float*)d_in[7];
  const float* w_res      = (const float*)d_in[8];
  const float* b_res      = (const float*)d_in[9];
  const float* w_res_last = (const float*)d_in[10];
  const float* b_res_last = (const float*)d_in[11];
  const float* w_end      = (const float*)d_in[12];
  const float* b_end      = (const float*)d_in[13];
  float* out = (float*)d_out;

  char* wp = (char*)d_ws;
  auto alloc = [&](size_t bytes) -> char* {
    char* p = wp;
    wp += (bytes + 255) & ~(size_t)255;
    return p;
  };
  __bf16* spect_bf   = (__bf16*)alloc((size_t)M_TOT * SPC * 2);
  float*  started_f  = (float*) alloc((size_t)M_TOT * NCH * 4);
  __bf16* started_bf = (__bf16*)alloc((size_t)M_TOT * NCH * 2);
  __bf16* act_bf     = (__bf16*)alloc((size_t)M_TOT * NCH * 2);
  float*  skip_f     = (float*) alloc((size_t)M_TOT * NCH * 4);
  __bf16* w_in_T     = (__bf16*)alloc((size_t)NL * 3 * 512 * NCH * 2);
  __bf16* w_cond_T   = (__bf16*)alloc((size_t)NL * 512 * SPC * 2);
  __bf16* w_res_T    = (__bf16*)alloc((size_t)(NL - 1) * 512 * NCH * 2);
  __bf16* w_rl_T     = (__bf16*)alloc((size_t)NCH * NCH * 2);
  float*  bias_g     = (float*) alloc((size_t)NL * 512 * 4);
  __bf16* zguard     = (__bf16*)alloc((size_t)NCH * 2);  // 512 B of zeros

  const int TB = 256;
  // --- prep ---
  hipMemsetAsync(zguard, 0, (size_t)NCH * 2, stream);
  {
    long n = (long)M_TOT * SPC;
    k_cvt_bf16<<<(unsigned)((n + TB - 1) / TB), TB, 0, stream>>>(spect, spect_bf, n);
  }
  {
    long n = (long)NL * 3 * 256 * 512;
    k_transpose_cvt<<<(unsigned)((n + TB - 1) / TB), TB, 0, stream>>>(
        w_in, w_in_T, NL * 3, 256, 512);
  }
  {
    long n = (long)NL * SPC * 512;
    k_transpose_cvt<<<(unsigned)((n + TB - 1) / TB), TB, 0, stream>>>(
        w_cond, w_cond_T, NL, SPC, 512);
  }
  {
    long n = (long)(NL - 1) * 256 * 512;
    k_transpose_cvt<<<(unsigned)((n + TB - 1) / TB), TB, 0, stream>>>(
        w_res, w_res_T, NL - 1, 256, 512);
  }
  {
    long n = 256L * 256;
    k_transpose_cvt<<<(unsigned)((n + TB - 1) / TB), TB, 0, stream>>>(
        w_res_last, w_rl_T, 1, 256, 256);
  }
  k_bias_add<<<(NL * 512 + TB - 1) / TB, TB, 0, stream>>>(b_in, b_cond, bias_g, NL * 512);
  k_start<<<(M_TOT * NCH + TB - 1) / TB, TB, 0, stream>>>(audio, w_start, b_start,
                                                          started_f, started_bf);

  // --- 12 WaveNet layers ---
  const int nblk = M_TOT / (16 * MSUB);  // 512 blocks of 64 rows, 8 waves = 512 cols
  for (int i = 0; i < NL; ++i) {
    k_gate<<<nblk, 256, 0, stream>>>(
        started_bf, spect_bf,
        w_in_T + (size_t)i * 3 * 512 * NCH,
        w_cond_T + (size_t)i * 512 * SPC,
        bias_g + (size_t)i * 512,
        zguard, act_bf, 1 << i);
    if (i < NL - 1) {
      k_res<<<nblk, 256, 0, stream>>>(
          act_bf, w_res_T + (size_t)i * 512 * NCH, b_res + (size_t)i * 512,
          started_f, started_bf, skip_f, (i == 0) ? 1 : 0);
    } else {
      k_last<<<nblk, 256, 0, stream>>>(act_bf, w_rl_T, b_res_last, skip_f);
    }
  }

  // --- final projection 256 -> 8 ---
  k_end<<<(M_TOT * 8 + TB - 1) / TB, TB, 0, stream>>>(skip_f, w_end, b_end, out);
}
